// MultiHeadAttention_70987219468989
// MI455X (gfx1250) — compile-verified, matrix operand order checked by reading, not run on hardware
//
#include <hip/hip_runtime.h>

#define BB 4
#define NN 2048
#define MM 2048
#define CC 256
#define HH 8
#define DD 32
#define PITCH 2052   // 2048 + 4-float pad to spread LDS banks across rows

typedef __attribute__((ext_vector_type(16))) __bf16          v16bf;
typedef __attribute__((ext_vector_type(8)))  float           v8f;
typedef __attribute__((ext_vector_type(16))) unsigned short  v16u;
typedef __attribute__((ext_vector_type(8)))  unsigned int    v8u;
typedef __attribute__((ext_vector_type(8)))  unsigned short  su8;

// ---- packed f32x2 -> bf16x2 -------------------------------------------------
__device__ __forceinline__ unsigned int pack_bf2(float lo, float hi) {
#if __has_builtin(__builtin_amdgcn_cvt_pk_bf16_f32)
  typedef __attribute__((ext_vector_type(2))) __bf16 v2bf;
  v2bf p = __builtin_amdgcn_cvt_pk_bf16_f32(lo, hi);
  return __builtin_bit_cast(unsigned int, p);
#else
  unsigned int a = __builtin_bit_cast(unsigned int, lo);
  unsigned int b = __builtin_bit_cast(unsigned int, hi);
  a += 0x7FFFu + ((a >> 16) & 1u);          // RNE
  b += 0x7FFFu + ((b >> 16) & 1u);
  // bytes of {b(hi dword):a(lo dword)}; take a[31:16] then b[31:16]
  return __builtin_amdgcn_perm(b, a, 0x07060302u);
#endif
}

__device__ __forceinline__ unsigned short f2bf(float f) {
  unsigned int u = __builtin_bit_cast(unsigned int, f);
  u += 0x7FFFu + ((u >> 16) & 1u);
  return (unsigned short)(u >> 16);
}

// Fragment from two contiguous 8 x bf16 runs (A/B 16-bit WMMA layout:
// per lane, elements 0..7 = K-run at (half*8), elements 8..15 = K-run at (16+half*8))
__device__ __forceinline__ v16bf frag_from_u16(su8 lo, su8 hi) {
  v16u u;
#pragma unroll
  for (int j = 0; j < 8; ++j) { u[j] = lo[j]; u[j + 8] = hi[j]; }
  return __builtin_bit_cast(v16bf, u);
}

// Same, but converting two contiguous 8 x f32 runs to bf16 (packed cvt)
__device__ __forceinline__ v16bf frag_from_f32(const float* p0, const float* p1) {
  float4 a0 = ((const float4*)p0)[0];
  float4 a1 = ((const float4*)p0)[1];
  float4 b0 = ((const float4*)p1)[0];
  float4 b1 = ((const float4*)p1)[1];
  v8u u;
  u[0] = pack_bf2(a0.x, a0.y); u[1] = pack_bf2(a0.z, a0.w);
  u[2] = pack_bf2(a1.x, a1.y); u[3] = pack_bf2(a1.z, a1.w);
  u[4] = pack_bf2(b0.x, b0.y); u[5] = pack_bf2(b0.z, b0.w);
  u[6] = pack_bf2(b1.x, b1.y); u[7] = pack_bf2(b1.z, b1.w);
  return __builtin_bit_cast(v16bf, u);
}

// -------------------------------------------------------------------------
// Kernel 0: bulk f32 -> bf16 conversion (one pass per tensor).
// -------------------------------------------------------------------------
__global__ void __launch_bounds__(256)
cvt_bf16_kernel(const float* __restrict__ src, unsigned short* __restrict__ dst,
                int n4) {
  const float4* s4 = (const float4*)src;
  for (int i = blockIdx.x * 256 + threadIdx.x; i < n4;
       i += gridDim.x * 256) {
    float4 v = s4[i];
    uint2 p;
    p.x = pack_bf2(v.x, v.y);
    p.y = pack_bf2(v.z, v.w);
    *(uint2*)(dst + (size_t)i * 4) = p;
  }
}

// -------------------------------------------------------------------------
// Kernel 1: fused dual-GEMM projections (all operands pre-converted bf16).
//   p=0: Qc = input_q@Wq.T + embedding_q@Wpq.T + bq+bpq   -> (B,H,N,32) bf16
//   p=1: Kc = input_k@Wk.T + embedding_k@Wpk.T + bk+bpk   -> (B,H,M,32) bf16
//   p=2: Vt = input_v@Wv.T + embedding_k@Wvk.T + bv+bvk   -> (B,H,32,M) bf16
// One wave per 16x16 output tile; K=256 in 8 steps of 32 with 2 WMMAs/step.
// -------------------------------------------------------------------------
__global__ void __launch_bounds__(256)
proj_kernel(const unsigned short* __restrict__ Xq, const unsigned short* __restrict__ Xk,
            const unsigned short* __restrict__ Xv, const unsigned short* __restrict__ Eq,
            const unsigned short* __restrict__ Ek,
            const unsigned short* __restrict__ Wq,  const float* __restrict__ bq,
            const unsigned short* __restrict__ Wk,  const float* __restrict__ bk,
            const unsigned short* __restrict__ Wv,  const float* __restrict__ bv,
            const unsigned short* __restrict__ Wpq, const float* __restrict__ bpq,
            const unsigned short* __restrict__ Wpk, const float* __restrict__ bpk,
            const unsigned short* __restrict__ Wvk, const float* __restrict__ bvk,
            unsigned short* __restrict__ Qc, unsigned short* __restrict__ Kc,
            unsigned short* __restrict__ Vt) {
  const int tid  = threadIdx.x;
  const int lane = tid & 31, wave = tid >> 5;
  const int half = lane >> 4, l16 = lane & 15;

  int job = blockIdx.x * 8 + wave;           // 24576 jobs total
  const int p  = job >> 13;                  // 0..2
  int rem = job & 8191;
  const int b  = rem >> 11;                  // 0..3
  rem &= 2047;
  const int rt = rem >> 4;                   // row tile 0..127
  const int ct = rem & 15;                   // col tile 0..15

  const unsigned short* X1 = (p == 0) ? Xq  : (p == 1) ? Xk  : Xv;
  const unsigned short* X2 = (p == 0) ? Eq  : Ek;
  const unsigned short* W1 = (p == 0) ? Wq  : (p == 1) ? Wk  : Wv;
  const unsigned short* W2 = (p == 0) ? Wpq : (p == 1) ? Wpk : Wvk;
  const float* B1 = (p == 0) ? bq  : (p == 1) ? bk  : bv;
  const float* B2 = (p == 0) ? bpq : (p == 1) ? bpk : bvk;

  const int n = rt * 16 + l16;               // A-frag row (token)
  const int c = ct * 16 + l16;               // B-frag column (out channel)
  const unsigned short* x1row = X1 + (size_t)(b * NN + n) * CC;
  const unsigned short* x2row = X2 + (size_t)(b * NN + n) * CC;
  const unsigned short* w1row = W1 + (size_t)c * CC;  // y = x@W.T => B[k][c] = W[c][k]
  const unsigned short* w2row = W2 + (size_t)c * CC;

  v8f acc = {};
#pragma unroll 4
  for (int kk = 0; kk < CC; kk += 32) {
    const int o0 = kk + half * 8;
    const int o1 = kk + 16 + half * 8;
    v16bf a2 = frag_from_u16(*(const su8*)(x2row + o0), *(const su8*)(x2row + o1));
    v16bf w2 = frag_from_u16(*(const su8*)(w2row + o0), *(const su8*)(w2row + o1));
    acc = __builtin_amdgcn_wmma_f32_16x16x32_bf16(false, a2, false, w2,
                                                  (short)0, acc, false, false);
    v16bf a1 = frag_from_u16(*(const su8*)(x1row + o0), *(const su8*)(x1row + o1));
    v16bf w1 = frag_from_u16(*(const su8*)(w1row + o0), *(const su8*)(w1row + o1));
    acc = __builtin_amdgcn_wmma_f32_16x16x32_bf16(false, a1, false, w1,
                                                  (short)0, acc, false, false);
  }

  const int h = c >> 5, d = c & 31;
  const float bias = B1[c] + B2[c];
#pragma unroll
  for (int i = 0; i < 8; ++i) {              // D layout: row = i + 8*half, col = l16
    const int ni = rt * 16 + i + 8 * half;
    const float val = acc[i] + bias;
    if (p == 2)
      Vt[(size_t)((b * HH + h) * DD + d) * MM + ni] = f2bf(val);
    else {
      unsigned short* out = (p == 0) ? Qc : Kc;
      out[(size_t)((b * HH + h) * NN + ni) * DD + d] = f2bf(val);
    }
  }
}

// -------------------------------------------------------------------------
// Kernel 2: fused attention. One block = one (b, h, 16-row stripe of N).
// LDS holds the full 16 x 2048 score stripe; raw scores never hit HBM.
//   pass1: scores = Q*K^T via WMMA; K-tiles staged global->LDS with
//          double-buffered global_load_async_to_lds_b128 (per-wave pipeline)
//   pass2: scale * factors * key_weights, masks -> -inf, row max  (128-bit IO)
//   pass3: exp, row sum (un-normalized e kept in LDS)
//   pass4: normalized attn streamed to HBM with global_store_b128
//   pass5: hidden = e @ (v+vk) via WMMA, ds_add_f32 reduce, scale by 1/rowsum
// -------------------------------------------------------------------------
#define SC_FLOATS   (16 * PITCH)
#define HID_FLOATS  512
#define RED_FLOATS  256
#define STAGE_BYTE_BASE ((SC_FLOATS + HID_FLOATS + RED_FLOATS) * 4)
#define STAGE_BYTES_PER_WAVE 2048          // 2 buffers x 1KB tile

__global__ void __launch_bounds__(256)
attn_kernel(const unsigned short* __restrict__ Qc,
            const unsigned short* __restrict__ Kc,
            const unsigned short* __restrict__ Vt,
            const float* __restrict__ kw,   // key_weights  (B,M)
            const int*   __restrict__ km,   // key_masks    (B,M)
            const float* __restrict__ af,   // attention_factors (B,N,M)
            const int*   __restrict__ am,   // attention_masks (B,1,N,M)
            float* __restrict__ outHidden,  // (B,N,C)
            float* __restrict__ outAttn) {  // (B,H,N,M)
  extern __shared__ float smem[];
  float* sc  = smem;                         // 16 * PITCH score stripe
  float* hid = smem + SC_FLOATS;             // 16 x 32 hidden reduction buffer
  float* red = hid + HID_FLOATS;             // 16 x 16 reduce scratch (+rinv)

  const int tid  = threadIdx.x;
  const int lane = tid & 31, wave = tid >> 5;
  const int half = lane >> 4, l16 = lane & 15;
  const int nt = blockIdx.x & 127;
  const int h  = (blockIdx.x >> 7) & 7;
  const int b  = blockIdx.x >> 10;

  hid[tid] = 0.f; hid[tid + 256] = 0.f;

  // ---- pass 1: raw QK^T into LDS (async-staged K tiles) ----
  const size_t qbase = ((size_t)((b * HH + h) * NN) + nt * 16 + l16) * DD;
  v16bf afrag = frag_from_u16(*(const su8*)(Qc + qbase + half * 8),
                              *(const su8*)(Qc + qbase + 16 + half * 8));

  const size_t headK = (size_t)((b * HH + h) * MM) * DD;  // Kc head slice (elems)
  const int stageOff = STAGE_BYTE_BASE + wave * STAGE_BYTES_PER_WAVE;
  const unsigned short* stage =
      (const unsigned short*)((const char*)smem + stageOff);

  // issue async copy of K tile j (1KB, contiguous) into staging buffer `buf`
  auto issue = [&](int j, int buf) {
    const unsigned short* g = Kc + headK + (size_t)j * 512 + lane * 8; // 16B/lane
    int lo = stageOff + buf * 1024 + lane * 16;
    asm volatile("global_load_async_to_lds_b128 %0, %1, off"
                 :: "v"(lo), "v"(g) : "memory");
    const unsigned short* g2 = g + 256;                                // +512B
    int lo2 = lo + 512;
    asm volatile("global_load_async_to_lds_b128 %0, %1, off"
                 :: "v"(lo2), "v"(g2) : "memory");
  };

  issue(wave, 0);                            // tile t=0
#pragma unroll 1
  for (int t = 0; t < 16; ++t) {             // tiles j = wave + 8*t
    if (t + 1 < 16) {
      issue(wave + 8 * (t + 1), (t + 1) & 1);
      asm volatile("s_wait_asynccnt 2" ::: "memory");  // batch t done
    } else {
      asm volatile("s_wait_asynccnt 0" ::: "memory");
    }
    const unsigned short* st = stage + (t & 1) * 512;  // 16x32 bf16 tile
    v16bf bfrag = frag_from_u16(*(const su8*)(st + l16 * 32 + half * 8),
                                *(const su8*)(st + l16 * 32 + 16 + half * 8));
    v8f acc = {};
    acc = __builtin_amdgcn_wmma_f32_16x16x32_bf16(false, afrag, false, bfrag,
                                                  (short)0, acc, false, false);
    const int j = wave + 8 * t;
#pragma unroll
    for (int i = 0; i < 8; ++i)
      sc[(i + 8 * half) * PITCH + j * 16 + l16] = acc[i];
  }
  __syncthreads();

  // ---- pass 2: scale/factor/weight/mask + row max (float4/int4 IO) ----
  const int r = tid >> 4, q = tid & 15;      // 16 threads per row, 4-elem chunks
  const int n_row = nt * 16 + r;
  const float scale = 0.17677669529663687f;  // 1/sqrt(32)
  const float* afrow = af + (size_t)(b * NN + n_row) * MM;
  const int*   amrow = am + (size_t)(b * NN + n_row) * MM;
  const float* kwrow = kw + (size_t)b * MM;
  const int*   kmrow = km + (size_t)b * MM;
  const float NEG_INF = -__builtin_huge_valf();

  float mx = NEG_INF;
  for (int s = 0; s < 32; ++s) {
    const int m = 4 * q + 64 * s;
    float4 sv = *(float4*)&sc[r * PITCH + m];
    const float4 afv = *(const float4*)(afrow + m);
    const int4   amv = *(const int4*)(amrow + m);
    const float4 kwv = *(const float4*)(kwrow + m);
    const int4   kmv = *(const int4*)(kmrow + m);
    sv.x = (kmv.x | amv.x) ? NEG_INF : sv.x * scale * afv.x * kwv.x;
    sv.y = (kmv.y | amv.y) ? NEG_INF : sv.y * scale * afv.y * kwv.y;
    sv.z = (kmv.z | amv.z) ? NEG_INF : sv.z * scale * afv.z * kwv.z;
    sv.w = (kmv.w | amv.w) ? NEG_INF : sv.w * scale * afv.w * kwv.w;
    *(float4*)&sc[r * PITCH + m] = sv;
    mx = fmaxf(mx, fmaxf(fmaxf(sv.x, sv.y), fmaxf(sv.z, sv.w)));
  }
  red[r * 16 + q] = mx;
  __syncthreads();
  float rowmax = red[r * 16];
#pragma unroll
  for (int t = 1; t < 16; ++t) rowmax = fmaxf(rowmax, red[r * 16 + t]);
  __syncthreads();

  // ---- pass 3: exp + row sum (e values stay un-normalized in LDS) ----
  float sum = 0.f;
  for (int s = 0; s < 32; ++s) {
    const int m = 4 * q + 64 * s;
    float4 sv = *(float4*)&sc[r * PITCH + m];
    sv.x = __expf(sv.x - rowmax);
    sv.y = __expf(sv.y - rowmax);
    sv.z = __expf(sv.z - rowmax);
    sv.w = __expf(sv.w - rowmax);
    *(float4*)&sc[r * PITCH + m] = sv;
    sum += (sv.x + sv.y) + (sv.z + sv.w);
  }
  red[r * 16 + q] = sum;
  __syncthreads();
  float rowsum = 0.f;
#pragma unroll
  for (int t = 0; t < 16; ++t) rowsum += red[r * 16 + t];
  const float rinv = 1.0f / rowsum;
  __syncthreads();                           // all reads of red done
  if (q == 0) red[r] = rinv;                 // stash per-row 1/sum for epilogue

  // ---- pass 4: stream normalized attn to HBM (b128 stores) ----
  float* attnrow = outAttn + ((size_t)((b * HH + h) * NN) + n_row) * MM;
  for (int s = 0; s < 32; ++s) {
    const int m = 4 * q + 64 * s;
    float4 sv = *(float4*)&sc[r * PITCH + m];
    sv.x *= rinv; sv.y *= rinv; sv.z *= rinv; sv.w *= rinv;
    *(float4*)(attnrow + m) = sv;
  }
  __syncthreads();

  // ---- pass 5: hidden = e @ (v+vk); each wave owns a 256-wide K slice ----
  v8f acc0 = {}, acc1 = {};
  const size_t vbase = (size_t)((b * HH + h) * DD) * MM;   // Vt is (32, M)
  for (int kt = 0; kt < 8; ++kt) {
    const int kk = wave * 256 + kt * 32;
    const float* p0 = sc + l16 * PITCH + kk + half * 8;
    v16bf aa = frag_from_f32(p0, p0 + 16);
    const unsigned short* v0 = Vt + vbase + (size_t)l16 * MM + kk + half * 8;
    v16bf b0 = frag_from_u16(*(const su8*)v0, *(const su8*)(v0 + 16));
    acc0 = __builtin_amdgcn_wmma_f32_16x16x32_bf16(false, aa, false, b0,
                                                   (short)0, acc0, false, false);
    const unsigned short* v1 = Vt + vbase + (size_t)(16 + l16) * MM + kk + half * 8;
    v16bf b1 = frag_from_u16(*(const su8*)v1, *(const su8*)(v1 + 16));
    acc1 = __builtin_amdgcn_wmma_f32_16x16x32_bf16(false, aa, false, b1,
                                                   (short)0, acc1, false, false);
  }
#pragma unroll
  for (int i = 0; i < 8; ++i) {              // ds_add_f32 cross-wave reduce
    atomicAdd(&hid[(i + 8 * half) * 32 + l16],      acc0[i]);
    atomicAdd(&hid[(i + 8 * half) * 32 + 16 + l16], acc1[i]);
  }
  __syncthreads();

  for (int idx = tid; idx < 512; idx += 256) {
    const int rr = idx >> 5, d = idx & 31;
    outHidden[(size_t)(b * NN + nt * 16 + rr) * CC + h * DD + d] =
        hid[idx] * red[rr];                  // fold 1/rowsum here
  }
}

// -------------------------------------------------------------------------
extern "C" void kernel_launch(void* const* d_in, const int* in_sizes, int n_in,
                              void* d_out, int out_size, void* d_ws, size_t ws_size,
                              hipStream_t stream) {
  const float* input_q     = (const float*)d_in[0];
  const float* input_k     = (const float*)d_in[1];
  const float* input_v     = (const float*)d_in[2];
  const float* embedding_q = (const float*)d_in[3];
  const float* embedding_k = (const float*)d_in[4];
  const float* key_weights = (const float*)d_in[5];
  const int*   key_masks   = (const int*)  d_in[6];
  const float* attn_factor = (const float*)d_in[7];
  const int*   attn_masks  = (const int*)  d_in[8];
  const float* Wq  = (const float*)d_in[9],  *bq  = (const float*)d_in[10];
  const float* Wk  = (const float*)d_in[11], *bk  = (const float*)d_in[12];
  const float* Wv  = (const float*)d_in[13], *bv  = (const float*)d_in[14];
  const float* Wpq = (const float*)d_in[15], *bpq = (const float*)d_in[16];
  const float* Wpk = (const float*)d_in[17], *bpk = (const float*)d_in[18];
  const float* Wvk = (const float*)d_in[19], *bvk = (const float*)d_in[20];

  const size_t NX = (size_t)BB * NN * CC;   // 2,097,152 elems per activation
  const size_t NW = (size_t)CC * CC;        // 65,536 elems per weight

  unsigned short* ws = (unsigned short*)d_ws;
  unsigned short* XqB  = ws;                 ws += NX;
  unsigned short* XkB  = ws;                 ws += NX;
  unsigned short* XvB  = ws;                 ws += NX;
  unsigned short* EqB  = ws;                 ws += NX;
  unsigned short* EkB  = ws;                 ws += NX;
  unsigned short* WqB  = ws;                 ws += NW;
  unsigned short* WkB  = ws;                 ws += NW;
  unsigned short* WvB  = ws;                 ws += NW;
  unsigned short* WpqB = ws;                 ws += NW;
  unsigned short* WpkB = ws;                 ws += NW;
  unsigned short* WvkB = ws;                 ws += NW;
  unsigned short* Qc   = ws;                 ws += NX;  // (B,H,N,32) bf16
  unsigned short* Kc   = ws;                 ws += NX;  // (B,H,M,32) bf16
  unsigned short* Vt   = ws;                 ws += NX;  // (B,H,32,M) bf16

  float* outHidden = (float*)d_out;                       // (B,N,C)
  float* outAttn   = outHidden + (size_t)BB * NN * CC;    // (B,H,N,M)

  // --- bulk f32 -> bf16 conversions (memory-bound pre-pass) ---
  const int nx4 = (int)(NX / 4), nw4 = (int)(NW / 4);
  cvt_bf16_kernel<<<2048, 256, 0, stream>>>(input_q,     XqB,  nx4);
  cvt_bf16_kernel<<<2048, 256, 0, stream>>>(input_k,     XkB,  nx4);
  cvt_bf16_kernel<<<2048, 256, 0, stream>>>(input_v,     XvB,  nx4);
  cvt_bf16_kernel<<<2048, 256, 0, stream>>>(embedding_q, EqB,  nx4);
  cvt_bf16_kernel<<<2048, 256, 0, stream>>>(embedding_k, EkB,  nx4);
  cvt_bf16_kernel<<<64,   256, 0, stream>>>(Wq,  WqB,  nw4);
  cvt_bf16_kernel<<<64,   256, 0, stream>>>(Wk,  WkB,  nw4);
  cvt_bf16_kernel<<<64,   256, 0, stream>>>(Wv,  WvB,  nw4);
  cvt_bf16_kernel<<<64,   256, 0, stream>>>(Wpq, WpqB, nw4);
  cvt_bf16_kernel<<<64,   256, 0, stream>>>(Wpk, WpkB, nw4);
  cvt_bf16_kernel<<<64,   256, 0, stream>>>(Wvk, WvkB, nw4);

  proj_kernel<<<3072, 256, 0, stream>>>(
      XqB, XkB, XvB, EqB, EkB,
      WqB, bq, WkB, bk, WvB, bv, WpqB, bpq, WpkB, bpk, WvkB, bvk,
      Qc, Kc, Vt);

  const size_t shmem =
      (size_t)STAGE_BYTE_BASE + 8 * STAGE_BYTES_PER_WAVE;  // ~150.8 KB
  attn_kernel<<<BB * HH * (NN / 16), 256, shmem, stream>>>(
      Qc, Kc, Vt, key_weights, key_masks, attn_factor, attn_masks,
      outHidden, outAttn);
}
